// NeuralPOSTagger_15006615733641
// MI455X (gfx1250) — matrix operand order repaired and checked by reading, not compiled, not run on hardware
//
#include <hip/hip_runtime.h>
#include <hip/hip_bf16.h>

// ---------------------------------------------------------------------------
// BiLSTM-CRF POS tagger forward pass for gfx1250 (MI455X), wave32 + WMMA.
// Persistent per-layer LSTM kernel: 32 resident blocks run all 256 timesteps
// with a device-scope barrier between steps (no per-step launch overhead).
// All heavy matmuls use v_wmma_f32_16x16x32_f16 (f16 in, f32 accumulate).
// ---------------------------------------------------------------------------

typedef _Float16 v16h  __attribute__((ext_vector_type(16)));
typedef _Float16 half8 __attribute__((ext_vector_type(8)));
typedef float    v8f   __attribute__((ext_vector_type(8)));

constexpr int E  = 256;
constexpr int H  = 512;
constexpr int T  = 18;
constexpr int B  = 128;
constexpr int S  = 256;
constexpr int H2 = 2 * H;     // 1024
constexpr int G4 = 4 * H;     // 2048

constexpr int LSTM_BLOCKS_X = 16;   // waves/dir = 128 : 4 batch-pair tiles x 32 col tiles
constexpr unsigned LSTM_NBLK = LSTM_BLOCKS_X * 2;

union Frag { v16h v; half8 h[2]; };

// A-matrix 16x32 f16 fragment: lane holds row (lane&15).
// halves 0..7 = K kb..kb+7 ; halves 8..15 = K 16+kb..16+kb+7 ; kb = (lane>=16)*8
__device__ __forceinline__ v16h load_a_frag(const _Float16* row, int lane) {
  int kb = (lane >> 4) << 3;
  Frag f;
  f.h[0] = *(const half8*)(row + kb);
  f.h[1] = *(const half8*)(row + 16 + kb);
  return f.v;
}

// B-matrix 32x16 f16 fragment: lane holds column (lane&15) = a row of W[N,K].
// halves 0..7 = K kb..kb+7 ; halves 8..15 = K 16+kb..16+kb+7 ; kb = (lane>=16)*8
__device__ __forceinline__ v16h load_b_frag(const _Float16* wrow, int lane) {
  int kb = (lane >> 4) << 3;
  Frag f;
  f.h[0] = *(const half8*)(wrow + kb);
  f.h[1] = *(const half8*)(wrow + 16 + kb);
  return f.v;
}

__device__ __forceinline__ v8f wmma_f16(v16h a, v16h b, v8f c) {
  return __builtin_amdgcn_wmma_f32_16x16x32_f16(
      /*neg_a=*/false, a, /*neg_b=*/false, b,
      /*c_mod=*/(short)0, c, /*reuse_a=*/false, /*reuse_b=*/false);
}

__device__ __forceinline__ float sigm(float x) { return 1.0f / (1.0f + __expf(-x)); }

// Device-scope split barrier across all resident blocks of the layer kernel.
// gen target = (t+1) * nblk; counter is zeroed by the host before launch.
__device__ __forceinline__ void grid_barrier(unsigned* bar, unsigned target) {
  __threadfence();          // release our global stores (h state)
  __syncthreads();
  if (threadIdx.x == 0) {
    __hip_atomic_fetch_add(bar, 1u, __ATOMIC_ACQ_REL, __HIP_MEMORY_SCOPE_AGENT);
    while (__hip_atomic_load(bar, __ATOMIC_ACQUIRE, __HIP_MEMORY_SCOPE_AGENT) < target)
      __builtin_amdgcn_s_sleep(1);
  }
  __syncthreads();
  __threadfence();          // acquire other blocks' stores
}

// ---------------------------------------------------------------------------
// Utility kernels
// ---------------------------------------------------------------------------
__global__ void cvt_f32_f16_kernel(const float* __restrict__ src,
                                   _Float16* __restrict__ dst, int n) {
  int i = blockIdx.x * blockDim.x + threadIdx.x;
  if (i < n) dst[i] = (_Float16)src[i];
}

__global__ void zero_u32_kernel(unsigned* __restrict__ p, int n) {
  int i = blockIdx.x * blockDim.x + threadIdx.x;
  if (i < n) p[i] = 0u;
}

// Embedding gather -> time-major f16 x0[(s*B+b)*E + e]; padding_idx 0 -> 0.
__global__ void embed_kernel(const int* __restrict__ ids,
                             const float* __restrict__ emb,
                             _Float16* __restrict__ x0) {
  int i = blockIdx.x * blockDim.x + threadIdx.x;   // over S*B*E = 8.4M
  int e  = i & (E - 1);
  int sb = i >> 8;            // E = 256
  int b  = sb & (B - 1);
  int s  = sb >> 7;           // B = 128
  int id = ids[b * S + s];
  float v = (id == 0) ? 0.0f : emb[(size_t)id * E + e];
  x0[i] = (_Float16)v;
}

// ---------------------------------------------------------------------------
// Persistent bidirectional LSTM layer.
// grid = (16, 2[dir]), block = 256 (8 waves). Wave -> 32(batch) x 16(hidden)
// tile: 2 batch sub-tiles x 4 gates = 8 f32 accumulators; each W fragment is
// reused for 2 WMMAs (halves per-step L2 weight traffic vs 16-row tiles).
// All 256 timesteps run inside the kernel with a grid barrier between steps.
// ---------------------------------------------------------------------------
struct LayerArgs {
  const _Float16* x;          // [S*B, ldx], row = t*B + b
  const _Float16* wih[2];     // [4H, ldx]
  const _Float16* whh[2];     // [4H, H]
  const float*    bias[2];    // [4H]  (b_ih + b_hh combined, f32)
  _Float16*       h0[2];      // [B, H] ping
  _Float16*       h1[2];      // [B, H] pong
  float*          cst[2];     // [B, H]
  _Float16*       hs;         // [S*B, 2H]  (fwd -> cols 0..H-1, bwd -> H..2H-1)
  unsigned*       bar;        // barrier counter (zeroed by host)
  int ldx;
};

__global__ __launch_bounds__(256) void lstm_layer_kernel(LayerArgs A) {
  const int dir   = blockIdx.y;
  const int lane  = threadIdx.x & 31;
  const int wave  = (blockIdx.x * blockDim.x + threadIdx.x) >> 5;  // 0..127
  const int tileB = wave >> 5;    // 0..3   (B/32)
  const int tileH = wave & 31;    // 0..31  (H/16)
  const int m0 = tileB * 32;
  const int j0 = tileH * 16;

  const _Float16* wih  = A.wih[dir];
  const _Float16* whh  = A.whh[dir];
  const float*    bias = A.bias[dir];
  float*          cst  = A.cst[dir];
  const int ldx = A.ldx;
  const int lr  = lane & 15;
  const int hi  = lane >> 4;
  const int col = j0 + lr;

  float bv[4];
#pragma unroll
  for (int g = 0; g < 4; ++g) bv[g] = bias[g * H + j0 + lr];

  for (int t = 0; t < S; ++t) {
    const int tx = dir ? (S - 1 - t) : t;
    const _Float16* hprev = (t & 1) ? A.h1[dir] : A.h0[dir];
    _Float16*       hcur  = (t & 1) ? A.h0[dir] : A.h1[dir];

    v8f acc[4][2];
#pragma unroll
    for (int g = 0; g < 4; ++g)
#pragma unroll
      for (int mt = 0; mt < 2; ++mt)
#pragma unroll
        for (int e = 0; e < 8; ++e) acc[g][mt][e] = bv[g];

    // ---- input projection: x[tx] @ Wih^T ----
    {
      const _Float16* ar0 = A.x + (size_t)(tx * B + m0 + lr) * ldx;
      const _Float16* ar1 = ar0 + (size_t)16 * ldx;
      for (int k = 0; k < ldx; k += 32) {
        v16h a0 = load_a_frag(ar0 + k, lane);
        v16h a1 = load_a_frag(ar1 + k, lane);
#pragma unroll
        for (int g = 0; g < 4; ++g) {
          v16h bf = load_b_frag(wih + (size_t)(g * H + j0 + lr) * ldx + k, lane);
          acc[g][0] = wmma_f16(a0, bf, acc[g][0]);
          acc[g][1] = wmma_f16(a1, bf, acc[g][1]);
        }
      }
    }
    // ---- recurrent projection: h_prev @ Whh^T ----
    {
      const _Float16* ar0 = hprev + (size_t)(m0 + lr) * H;
      const _Float16* ar1 = ar0 + (size_t)16 * H;
      for (int k = 0; k < H; k += 32) {
        v16h a0 = load_a_frag(ar0 + k, lane);
        v16h a1 = load_a_frag(ar1 + k, lane);
#pragma unroll
        for (int g = 0; g < 4; ++g) {
          v16h bf = load_b_frag(whh + (size_t)(g * H + j0 + lr) * H + k, lane);
          acc[g][0] = wmma_f16(a0, bf, acc[g][0]);
          acc[g][1] = wmma_f16(a1, bf, acc[g][1]);
        }
      }
    }

    // ---- gate math. C-layout: lane l, elem e -> row = e + 8*(l>=16), col = l&15
#pragma unroll
    for (int mt = 0; mt < 2; ++mt) {
#pragma unroll
      for (int e = 0; e < 8; ++e) {
        const int row = m0 + mt * 16 + e + 8 * hi;
        float gi = sigm(acc[0][mt][e]);
        float gf = sigm(acc[1][mt][e]);
        float gg = tanhf(acc[2][mt][e]);
        float go = sigm(acc[3][mt][e]);
        float cold = cst[row * H + col];
        float cnew = gf * cold + gi * gg;
        cst[row * H + col] = cnew;
        float hv = go * tanhf(cnew);
        hcur[row * H + col] = (_Float16)hv;
        A.hs[(size_t)(tx * B + row) * H2 + dir * H + col] = (_Float16)hv;
      }
    }

    grid_barrier(A.bar, (unsigned)(t + 1) * LSTM_NBLK);
  }
}

// ---------------------------------------------------------------------------
// C[M,N] = relu?( A[M,K](f16) @ W[N,K]^T(f16) + bias ) -> f16. One wave/tile.
// ---------------------------------------------------------------------------
__global__ __launch_bounds__(256) void gemm_bias_relu_kernel(
    const _Float16* __restrict__ Am, const _Float16* __restrict__ Wm,
    const float* __restrict__ bias, _Float16* __restrict__ out,
    int M, int N, int K, int relu) {
  const int lane = threadIdx.x & 31;
  const int wave = (blockIdx.x * blockDim.x + threadIdx.x) >> 5;
  const int nt = N >> 4;
  if (wave >= (M >> 4) * nt) return;
  const int m0 = (wave / nt) * 16;
  const int n0 = (wave % nt) * 16;
  const int lr = lane & 15;

  v8f acc;
  float bv = bias[n0 + lr];
#pragma unroll
  for (int e = 0; e < 8; ++e) acc[e] = bv;

  const _Float16* arow = Am + (size_t)(m0 + lr) * K;
  const _Float16* wrow = Wm + (size_t)(n0 + lr) * K;
  for (int k = 0; k < K; k += 32)
    acc = wmma_f16(load_a_frag(arow + k, lane), load_b_frag(wrow + k, lane), acc);

  const int hi = lane >> 4;
#pragma unroll
  for (int e = 0; e < 8; ++e) {
    int row = m0 + e + 8 * hi;
    float v = acc[e];
    if (relu) v = fmaxf(v, 0.0f);
    out[(size_t)row * N + n0 + lr] = (_Float16)v;
  }
}

// ---------------------------------------------------------------------------
// Emissions: em[(s*B+b)*T + t] = fc_row . cls_w[t] + cls_b[t]. Wave per row,
// lane = tag (T=18 active lanes); fc row load broadcasts across lanes.
// ---------------------------------------------------------------------------
__global__ __launch_bounds__(128) void emissions_kernel(
    const _Float16* __restrict__ fc, const float* __restrict__ clsw,
    const float* __restrict__ clsb, float* __restrict__ em) {
  const int lane = threadIdx.x & 31;
  const int row  = (blockIdx.x * blockDim.x + threadIdx.x) >> 5;
  if (row >= S * B || lane >= T) return;
  const _Float16* frow = fc + (size_t)row * H;
  const float*    wrow = clsw + lane * H;
  float acc = 0.0f;
  for (int k = 0; k < H; k += 4) {
    float4 w4 = *(const float4*)(wrow + k);
    acc += (float)frow[k]     * w4.x + (float)frow[k + 1] * w4.y +
           (float)frow[k + 2] * w4.z + (float)frow[k + 3] * w4.w;
  }
  em[(size_t)row * T + lane] = acc + clsb[lane];
}

// ---------------------------------------------------------------------------
// CRF negative log-likelihood pieces. One wave per batch element; lane = tag
// for the forward-algorithm scan; lane 0 does the gold-path score.
// ---------------------------------------------------------------------------
__global__ __launch_bounds__(32) void crf_kernel(
    const int* __restrict__ ids, const int* __restrict__ tags,
    const float* __restrict__ em, const float* __restrict__ trans,
    const float* __restrict__ st, const float* __restrict__ et,
    float* __restrict__ llh) {
  const int b = blockIdx.x;
  const int lane = threadIdx.x;
  __shared__ float sT[T * T];
  __shared__ float sA[T];
  for (int i = lane; i < T * T; i += 32) sT[i] = trans[i];
  __syncthreads();
  if (lane < T) sA[lane] = st[lane] + em[(size_t)b * T + lane];
  __syncthreads();
  for (int s = 1; s < S; ++s) {
    bool mk = ids[b * S + s] != 0;
    float ns = 0.0f;
    if (lane < T) {
      float mx = -1e30f;
      for (int t = 0; t < T; ++t) mx = fmaxf(mx, sA[t] + sT[t * T + lane]);
      float sm = 0.0f;
      for (int t = 0; t < T; ++t) sm += __expf(sA[t] + sT[t * T + lane] - mx);
      ns = mx + __logf(sm) + em[((size_t)s * B + b) * T + lane];
    }
    __syncthreads();
    if (lane < T && mk) sA[lane] = ns;
    __syncthreads();
  }
  if (lane == 0) {
    float mx = -1e30f;
    for (int t = 0; t < T; ++t) mx = fmaxf(mx, sA[t] + et[t]);
    float sm = 0.0f;
    for (int t = 0; t < T; ++t) sm += __expf(sA[t] + et[t] - mx);
    float den = mx + __logf(sm);
    int tag0 = tags[b * S];
    float score = st[tag0] + em[(size_t)b * T + tag0];
    int cnt = (ids[b * S] != 0) ? 1 : 0;
    for (int s = 1; s < S; ++s) {
      if (ids[b * S + s] != 0) {
        int pv = tags[b * S + s - 1];
        int cu = tags[b * S + s];
        score += sT[pv * T + cu] + em[((size_t)s * B + b) * T + cu];
        cnt++;
      }
    }
    int se = cnt - 1;
    if (se < 0) se += S;             // emulate JAX negative-index wrap
    llh[b] = score + et[tags[b * S + se]] - den;
  }
}

__global__ __launch_bounds__(128) void mean_kernel(const float* __restrict__ llh,
                                                   float* __restrict__ out) {
  __shared__ float s[128];
  int i = threadIdx.x;
  s[i] = llh[i];
  __syncthreads();
  for (int k = 64; k > 0; k >>= 1) {
    if (i < k) s[i] += s[i + k];
    __syncthreads();
  }
  if (i == 0) out[0] = s[0] / (float)B;
}

// ---------------------------------------------------------------------------
// Host driver
// ---------------------------------------------------------------------------
extern "C" void kernel_launch(void* const* d_in, const int* in_sizes, int n_in,
                              void* d_out, int out_size, void* d_ws, size_t ws_size,
                              hipStream_t stream) {
  (void)in_sizes; (void)n_in; (void)out_size; (void)ws_size;
  const int*   ids         = (const int*)d_in[0];
  const int*   tags        = (const int*)d_in[1];
  const float* emb         = (const float*)d_in[2];
  const float* w_ih_l0f    = (const float*)d_in[3];
  const float* w_hh_l0f    = (const float*)d_in[4];
  const float* b_l0f       = (const float*)d_in[5];
  const float* w_ih_l0b    = (const float*)d_in[6];
  const float* w_hh_l0b    = (const float*)d_in[7];
  const float* b_l0b       = (const float*)d_in[8];
  const float* w_ih_l1f    = (const float*)d_in[9];
  const float* w_hh_l1f    = (const float*)d_in[10];
  const float* b_l1f       = (const float*)d_in[11];
  const float* w_ih_l1b    = (const float*)d_in[12];
  const float* w_hh_l1b    = (const float*)d_in[13];
  const float* b_l1b       = (const float*)d_in[14];
  const float* fc_w        = (const float*)d_in[15];
  const float* fc_b        = (const float*)d_in[16];
  const float* cls_w       = (const float*)d_in[17];
  const float* cls_b       = (const float*)d_in[18];
  const float* transitions = (const float*)d_in[19];
  const float* start_trans = (const float*)d_in[20];
  const float* end_trans   = (const float*)d_in[21];

  char* base = (char*)d_ws;
  size_t off = 0;
  auto alloc = [&](size_t bytes) -> void* {
    void* p = base + off;
    off += (bytes + 255) & ~(size_t)255;
    return p;
  };

  _Float16* x0h   = (_Float16*)alloc((size_t)S * B * E  * 2);
  _Float16* hs0   = (_Float16*)alloc((size_t)S * B * H2 * 2);
  _Float16* hs1   = (_Float16*)alloc((size_t)S * B * H2 * 2);
  _Float16* fco   = (_Float16*)alloc((size_t)S * B * H  * 2);
  float*    em    = (float*)   alloc((size_t)S * B * T  * 4);
  _Float16* wih0f = (_Float16*)alloc((size_t)G4 * E  * 2);
  _Float16* wih0b = (_Float16*)alloc((size_t)G4 * E  * 2);
  _Float16* whh0f = (_Float16*)alloc((size_t)G4 * H  * 2);
  _Float16* whh0b = (_Float16*)alloc((size_t)G4 * H  * 2);
  _Float16* wih1f = (_Float16*)alloc((size_t)G4 * H2 * 2);
  _Float16* wih1b = (_Float16*)alloc((size_t)G4 * H2 * 2);
  _Float16* whh1f = (_Float16*)alloc((size_t)G4 * H  * 2);
  _Float16* whh1b = (_Float16*)alloc((size_t)G4 * H  * 2);
  _Float16* fcw   = (_Float16*)alloc((size_t)H  * H2 * 2);
  float*    cst   = (float*)   alloc((size_t)2 * B * H * 4);   // [dir][B*H]
  _Float16* hbuf  = (_Float16*)alloc((size_t)4 * B * H * 2);   // [dir][pp][B*H]
  float*    llh   = (float*)   alloc((size_t)B * 4);
  unsigned* bar   = (unsigned*)alloc(256);

  auto cvt = [&](const float* s, _Float16* d, int n) {
    cvt_f32_f16_kernel<<<dim3((n + 255) / 256), dim3(256), 0, stream>>>(s, d, n);
  };
  cvt(w_ih_l0f, wih0f, G4 * E);
  cvt(w_ih_l0b, wih0b, G4 * E);
  cvt(w_hh_l0f, whh0f, G4 * H);
  cvt(w_hh_l0b, whh0b, G4 * H);
  cvt(w_ih_l1f, wih1f, G4 * H2);
  cvt(w_ih_l1b, wih1b, G4 * H2);
  cvt(w_hh_l1f, whh1f, G4 * H);
  cvt(w_hh_l1b, whh1b, G4 * H);
  cvt(fc_w,     fcw,   H * H2);

  {
    int n = S * B * E;
    embed_kernel<<<dim3(n / 256), dim3(256), 0, stream>>>(ids, emb, x0h);
  }

  const int BH = B * H;
  auto run_layer = [&](const _Float16* x, int ldx,
                       const _Float16* wf, const _Float16* whf, const float* bf,
                       const _Float16* wb, const _Float16* whb, const float* bb,
                       _Float16* hs_out) {
    zero_u32_kernel<<<dim3((2 * BH + 255) / 256), dim3(256), 0, stream>>>(
        (unsigned*)cst, 2 * BH);
    zero_u32_kernel<<<dim3((2 * BH + 255) / 256), dim3(256), 0, stream>>>(
        (unsigned*)hbuf, 2 * BH);   // 4*BH halves == 2*BH dwords
    zero_u32_kernel<<<dim3(1), dim3(64), 0, stream>>>(bar, 64);
    LayerArgs a;
    a.x = x; a.ldx = ldx; a.hs = hs_out; a.bar = bar;
    a.wih[0] = wf;  a.wih[1] = wb;
    a.whh[0] = whf; a.whh[1] = whb;
    a.bias[0] = bf; a.bias[1] = bb;
    a.h0[0] = hbuf + (size_t)0 * BH;  a.h1[0] = hbuf + (size_t)1 * BH;
    a.h0[1] = hbuf + (size_t)2 * BH;  a.h1[1] = hbuf + (size_t)3 * BH;
    a.cst[0] = cst; a.cst[1] = cst + BH;
    lstm_layer_kernel<<<dim3(LSTM_BLOCKS_X, 2, 1), dim3(256), 0, stream>>>(a);
  };

  run_layer(x0h, E,  wih0f, whh0f, b_l0f, wih0b, whh0b, b_l0b, hs0);
  run_layer(hs0, H2, wih1f, whh1f, b_l1f, wih1b, whh1b, b_l1b, hs1);

  {
    int M = S * B, N = H, K = H2;
    int waves = (M / 16) * (N / 16);
    gemm_bias_relu_kernel<<<dim3((waves + 7) / 8), dim3(256), 0, stream>>>(
        hs1, fcw, fc_b, fco, M, N, K, 1);
  }

  emissions_kernel<<<dim3(S * B / 4), dim3(128), 0, stream>>>(fco, cls_w, cls_b, em);
  crf_kernel<<<dim3(B), dim3(32), 0, stream>>>(ids, tags, em, transitions,
                                               start_trans, end_trans, llh);
  mean_kernel<<<dim3(1), dim3(128), 0, stream>>>(llh, (float*)d_out);
}
